// FrequencyFeatureExtractor_87265145520173
// MI455X (gfx1250) — matrix operand order repaired and checked by reading, not compiled, not run on hardware
//
#include <hip/hip_runtime.h>

// ---------------------------------------------------------------------------
// CDNA5 (gfx1250, wave32) pipeline:
//   k_gray   : [32,3,512,512] f32 -> grayscale bf16 G[16384][512]
//   k_basis1 : B1t[128][512] bf16   (rows 0..63 = cos, 64..127 = -sin) [c][w]
//   k_basis2 : A2 [64][1024] bf16   (h<512 = cos, h>=512 = sin)        row-major
//   k_wconv  : fc_w f32 -> Wb[256][4096] bf16 (K-contiguous per output col)
//   k_gemm1  : Zt[b][c][h]  = (G x B1t^T); wave owns a 16x128 strip.
//              B k-slice staged in LDS (double-buffered), A-fragment reused
//              across 8 accumulators, B fragments via ds_load_b128.
//   k_gemm2  : Xr[b][u][v]  = A2 x [Zr;Zi]; wave owns a 16x64 strip
//   k_gemm3  : out[32][256] = Xr x Wb^T + bias (f32 out)
// All GEMMs: v_wmma_f32_16x16x32_bf16 (bf16 in, f32 accumulate).
// ---------------------------------------------------------------------------

typedef __bf16 bf16_t;
typedef bf16_t v4bf  __attribute__((ext_vector_type(4)));
typedef bf16_t v8bf  __attribute__((ext_vector_type(8)));
typedef bf16_t v16bf __attribute__((ext_vector_type(16)));
typedef float  v4f   __attribute__((ext_vector_type(4)));
typedef float  v8f   __attribute__((ext_vector_type(8)));

#define CAT16(lo, hi) __builtin_shufflevector(lo, hi, 0,1,2,3,4,5,6,7,8,9,10,11,12,13,14,15)

#define TWO_PI_OVER_512 0.01227184630308513f

#define LDS_PITCH 40   // 32 payload + 8 pad bf16 => 80 B rows, 16B-aligned, conflict-free

static __device__ __forceinline__ bf16_t f2bf(float f) { return (bf16_t)f; }

// A-fragment 16x32 bf16 from row-major [row][k], leading dim ld.
// Contiguous k order (matches B): lane L holds row M=L&15; K = kb..kb+15, kb = k0 + 16*(L>=16).
static __device__ __forceinline__ v16bf frag_a(const bf16_t* __restrict__ base, int ld,
                                               int row0, int k0, int lane) {
  const bf16_t* p = base + (size_t)(row0 + (lane & 15)) * ld + (k0 + ((lane >> 4) << 4));
  v8bf lo = *(const v8bf*)p;
  v8bf hi = *(const v8bf*)(p + 8);
  return CAT16(lo, hi);
}

// B-fragment 32x16 bf16 from K-contiguous (transposed) storage base[n][k], leading dim ld.
// Contiguous k order (matches A): lane L holds col N=L&15; K = kb..kb+15, kb = k0 + 16*(L>=16).
static __device__ __forceinline__ v16bf frag_bt(const bf16_t* __restrict__ base, int ld,
                                                int col0, int k0, int lane) {
  const bf16_t* p = base + (size_t)(col0 + (lane & 15)) * ld + (k0 + ((lane >> 4) << 4));
  v8bf lo = *(const v8bf*)p;
  v8bf hi = *(const v8bf*)(p + 8);
  return CAT16(lo, hi);
}

// B-fragment from an LDS tile [128][LDS_PITCH] holding B1t[:, k0:k0+32].
static __device__ __forceinline__ v16bf frag_b_lds(const bf16_t* t, int n, int lane) {
  const bf16_t* p = t + ((n << 4) + (lane & 15)) * LDS_PITCH + ((lane >> 4) << 4);
  v8bf lo = *(const v8bf*)p;
  v8bf hi = *(const v8bf*)(p + 8);
  return CAT16(lo, hi);
}

// Cooperative copy of B1t[:, k0:k0+32] (128 rows) into an LDS tile. 256 threads,
// 16 bf16 (two 16B vectors) per thread.
static __device__ __forceinline__ void copy_b_tile(bf16_t* dst, const bf16_t* __restrict__ B1t,
                                                   int k0, int tid) {
  int row = tid >> 1;
  int cw  = (tid & 1) << 4;
  const uint4* s = (const uint4*)(B1t + (size_t)row * 512 + k0 + cw);
  uint4* d = (uint4*)(dst + row * LDS_PITCH + cw);
  d[0] = s[0];
  d[1] = s[1];
}

// --------------------------- prep kernels ----------------------------------

__global__ void k_gray(const float* __restrict__ x, bf16_t* __restrict__ g) {
  int t = blockIdx.x * blockDim.x + threadIdx.x;   // 4 pixels / thread
  int p = t << 2;                                  // pixel index in [0, 32*2^18)
  int b  = p >> 18;                                // 512*512 = 2^18
  int hw = p & 0x3FFFF;
  size_t base = ((size_t)b * 3) << 18;
  v4f r  = *(const v4f*)(x + base + hw);
  v4f gc = *(const v4f*)(x + base + (1 << 18) + hw);
  v4f bc = *(const v4f*)(x + base + (2 << 18) + hw);
  v4bf o;
#pragma unroll
  for (int i = 0; i < 4; ++i)
    o[i] = f2bf(0.299f * r[i] + 0.587f * gc[i] + 0.114f * bc[i]);
  *(v4bf*)(g + p) = o;
}

__global__ void k_basis1(bf16_t* __restrict__ b1t) {      // [128][512], [c][w]
  int t = blockIdx.x * blockDim.x + threadIdx.x;          // 65536
  int c = t >> 9, w = t & 511;
  int v = c & 63;
  float ang = (float)((v * w) & 511) * TWO_PI_OVER_512;
  float val = (c < 64) ? __cosf(ang) : -__sinf(ang);
  b1t[t] = f2bf(val);
}

__global__ void k_basis2(bf16_t* __restrict__ a2) {       // [64][1024], row-major
  int t = blockIdx.x * blockDim.x + threadIdx.x;          // 65536
  int u = t >> 10, hh = t & 1023;
  int h = hh & 511;
  float ang = (float)((u * h) & 511) * TWO_PI_OVER_512;
  float val = (hh < 512) ? __cosf(ang) : __sinf(ang);
  a2[t] = f2bf(val);
}

__global__ void k_wconv(const float* __restrict__ w, bf16_t* __restrict__ wb) {
  int t = blockIdx.x * blockDim.x + threadIdx.x;          // 262144, 4 elems each
  int p = t << 2;
  v4f v = *(const v4f*)(w + p);
  v4bf o;
#pragma unroll
  for (int i = 0; i < 4; ++i) o[i] = f2bf(v[i]);
  *(v4bf*)(wb + p) = o;
}

// --------------------------- WMMA GEMM stages ------------------------------

// Stage 1: Z = G[16384x512] x B1t^T[512x128]; write transposed Zt[b][c][h].
// Block = 8 waves = 128 rows; LDS double-buffers the 128x32 B k-slice.
__global__ void k_gemm1(const bf16_t* __restrict__ G, const bf16_t* __restrict__ B1t,
                        bf16_t* __restrict__ Zt) {
  __shared__ bf16_t lb[2][128 * LDS_PITCH];   // 2 x 10 KiB
  int tid  = threadIdx.x;
  int lane = tid & 31, wave = tid >> 5;
  int row0 = (blockIdx.x * 8 + wave) << 4;    // 16-row strip per wave

  copy_b_tile(lb[0], B1t, 0, tid);
  v8f acc[8] = {};
  v16bf a = frag_a(G, 512, row0, 0, lane);
  __syncthreads();

  for (int ks = 0; ks < 16; ++ks) {
    int buf = ks & 1;
    int kn  = (ks < 15) ? ((ks + 1) << 5) : 0;      // next slice (dummy 0 on last)
    copy_b_tile(lb[buf ^ 1], B1t, kn, tid);         // prefetch next B slice
    v16bf an = frag_a(G, 512, row0, kn, lane);      // prefetch next A fragment
    v16bf b[8];
#pragma unroll
    for (int n = 0; n < 8; ++n) b[n] = frag_b_lds(lb[buf], n, lane);
#pragma unroll
    for (int n = 0; n < 8; ++n)
      acc[n] = __builtin_amdgcn_wmma_f32_16x16x32_bf16(false, a, false, b[n], (short)0, acc[n],
                                                       false, false);
    a = an;
    __syncthreads();
  }

  // D layout: lane holds col N=(lane&15), rows row0 + 8*(lane>=16) + r.
  int bb = row0 >> 9;                       // batch
  int h0 = (row0 & 511) + ((lane >> 4) << 3);
#pragma unroll
  for (int n = 0; n < 8; ++n) {
    int c = (n << 4) + (lane & 15);
    v8bf o;
#pragma unroll
    for (int r = 0; r < 8; ++r) o[r] = f2bf(acc[n][r]);
    *(v8bf*)(Zt + ((((size_t)bb << 7) + c) << 9) + h0) = o;   // Zt[b][c][h0..h0+7]
  }
}

// Stage 2: Xr[b][64][64] = A2[64x1024] x [Zr;Zi][1024x64] (per batch).
// Each wave computes a 16x64 strip; B fragments batched to keep loads in flight.
__global__ void k_gemm2(const bf16_t* __restrict__ Zt, const bf16_t* __restrict__ A2,
                        bf16_t* __restrict__ Xr) {
  int lane = threadIdx.x & 31, wave = threadIdx.x >> 5;
  int t = blockIdx.x * 8 + wave;        // 128 strips: 32 batches x 4 u-tiles
  int b  = t >> 2;
  int u0 = (t & 3) << 4;
  const bf16_t* Zb = Zt + ((size_t)b << 16);   // 128*512 per batch
  v8f acc[4] = {};
#pragma unroll 4
  for (int k0 = 0; k0 < 1024; k0 += 32) {
    v16bf a = frag_a(A2, 1024, u0, k0, lane);
    int kb   = k0 + ((lane >> 4) << 4);
    int cimg = (kb < 512) ? 0 : 64;          // real cols 0..63, imag cols 64..127
    int kk   = kb & 511;
    v16bf bfrag[4];
#pragma unroll
    for (int n = 0; n < 4; ++n) {
      int col = (n << 4) + (lane & 15) + cimg;
      const bf16_t* p = Zb + ((size_t)col << 9) + kk;
      v8bf lo = *(const v8bf*)p;
      v8bf hi = *(const v8bf*)(p + 8);
      bfrag[n] = CAT16(lo, hi);
    }
#pragma unroll
    for (int n = 0; n < 4; ++n)
      acc[n] = __builtin_amdgcn_wmma_f32_16x16x32_bf16(false, a, false, bfrag[n], (short)0, acc[n],
                                                       false, false);
  }
  bf16_t* X = Xr + ((size_t)b << 12);          // [64][64] row-major
  int ub = u0 + ((lane >> 4) << 3);
#pragma unroll
  for (int n = 0; n < 4; ++n) {
    int v = (n << 4) + (lane & 15);
#pragma unroll
    for (int r = 0; r < 8; ++r) X[((ub + r) << 6) + v] = f2bf(acc[n][r]);
  }
}

// Stage 3: out[32][256] = Xr[32x4096] x Wb^T + bias  (f32 out).
__global__ void k_gemm3(const bf16_t* __restrict__ Xr, const bf16_t* __restrict__ Wb,
                        const float* __restrict__ bias, float* __restrict__ out) {
  int lane = threadIdx.x & 31, wave = threadIdx.x >> 5;
  int t  = blockIdx.x * 8 + wave;       // 32 tiles: 2 (M) x 16 (N)
  int tn = t & 15, tm = t >> 4;
  int m0 = tm << 4, n0 = tn << 4;
  v8f acc = {};
#pragma unroll 8
  for (int k0 = 0; k0 < 4096; k0 += 32) {
    v16bf a = frag_a(Xr, 4096, m0, k0, lane);
    v16bf b = frag_bt(Wb, 4096, n0, k0, lane);
    acc = __builtin_amdgcn_wmma_f32_16x16x32_bf16(false, a, false, b, (short)0, acc, false, false);
  }
  int n  = n0 + (lane & 15);
  int mb = m0 + ((lane >> 4) << 3);
  float bv = bias[n];
#pragma unroll
  for (int r = 0; r < 8; ++r) out[((mb + r) << 8) + n] = acc[r] + bv;
}

// --------------------------- launcher --------------------------------------

extern "C" void kernel_launch(void* const* d_in, const int* in_sizes, int n_in,
                              void* d_out, int out_size, void* d_ws, size_t ws_size,
                              hipStream_t stream) {
  const float* x    = (const float*)d_in[0];
  const float* fc_w = (const float*)d_in[1];
  const float* fc_b = (const float*)d_in[2];
  float* out = (float*)d_out;

  char* ws = (char*)d_ws;
  constexpr size_t OFF_G   = 0;                         // 16384*512*2  = 16 MiB
  constexpr size_t OFF_B1T = OFF_G   + (size_t)16384 * 512 * 2;
  constexpr size_t OFF_A2  = OFF_B1T + (size_t)128 * 512 * 2;
  constexpr size_t OFF_ZT  = OFF_A2  + (size_t)64 * 1024 * 2;
  constexpr size_t OFF_XR  = OFF_ZT  + (size_t)32 * 128 * 512 * 2;
  constexpr size_t OFF_WB  = OFF_XR  + (size_t)32 * 4096 * 2;

  bf16_t* G   = (bf16_t*)(ws + OFF_G);
  bf16_t* B1t = (bf16_t*)(ws + OFF_B1T);
  bf16_t* A2  = (bf16_t*)(ws + OFF_A2);
  bf16_t* Zt  = (bf16_t*)(ws + OFF_ZT);
  bf16_t* Xr  = (bf16_t*)(ws + OFF_XR);
  bf16_t* Wb  = (bf16_t*)(ws + OFF_WB);

  k_gray  <<<8192, 256, 0, stream>>>(x, G);             // 8.4M px / 4 per thread
  k_basis1<<< 256, 256, 0, stream>>>(B1t);
  k_basis2<<< 256, 256, 0, stream>>>(A2);
  k_wconv <<<1024, 256, 0, stream>>>(fc_w, Wb);

  k_gemm1 <<< 128, 256, 0, stream>>>(G, B1t, Zt);       // 1024 strips (16x128), 8 waves/block
  k_gemm2 <<<  16, 256, 0, stream>>>(Zt, A2, Xr);       // 128 strips (16x64)
  k_gemm3 <<<   4, 256, 0, stream>>>(Xr, Wb, fc_b, out);// 32 tiles
}
